// GraphAttentionLayer_12111807774825
// MI455X (gfx1250) — matrix operand order repaired and checked by reading, not compiled, not run on hardware
//
#include <hip/hip_runtime.h>
#include <hip/hip_bf16.h>

#define B_ 8
#define N_ 2048
#define D_ 128
#define ALPHA 0.2f

typedef __attribute__((ext_vector_type(16))) _Float16 v16h;
typedef __attribute__((ext_vector_type(8)))  _Float16 v8h;
typedef __attribute__((ext_vector_type(4)))  _Float16 v4h;
typedef __attribute__((ext_vector_type(8)))  float    v8f;

// ---------------------------------------------------------------------------
// Kernel 1: per-row scores f1 = h.a1, f2 = h.a2   (one wave32 per row)
// ---------------------------------------------------------------------------
__global__ __launch_bounds__(128) void gat_f1f2(const float* __restrict__ h,
                                                const float* __restrict__ a1,
                                                const float* __restrict__ a2,
                                                float* __restrict__ f1,
                                                float* __restrict__ f2) {
    int lane = threadIdx.x & 31;
    int wave = threadIdx.x >> 5;
    int row  = blockIdx.x * 4 + wave;            // row in [0, B*N)
    const float* hr = h + (size_t)row * D_;
    float s1 = 0.f, s2 = 0.f;
#pragma unroll
    for (int k = lane; k < D_; k += 32) {
        float v = hr[k];
        s1 += v * a1[k];
        s2 += v * a2[k];
    }
#pragma unroll
    for (int off = 16; off; off >>= 1) {
        s1 += __shfl_xor(s1, off);
        s2 += __shfl_xor(s2, off);
    }
    if (lane == 0) { f1[row] = s1; f2[row] = s2; }
}

// ---------------------------------------------------------------------------
// Kernel 2: h [B,N,D] f32 row-major  ->  h16T [B,D,N] f16 (transposed)
// ---------------------------------------------------------------------------
__global__ __launch_bounds__(256) void gat_transpose(const float* __restrict__ h,
                                                     _Float16* __restrict__ hT) {
    __shared__ float tile[32][33];
    int j0 = blockIdx.x * 32;
    int d0 = blockIdx.y * 32;
    int b  = blockIdx.z;
    int x = threadIdx.x;          // 0..31
    int y = threadIdx.y;          // 0..7
    const float* hb = h + (size_t)b * N_ * D_;
#pragma unroll
    for (int k = 0; k < 4; ++k) {
        int j = y + 8 * k;
        tile[j][x] = hb[(size_t)(j0 + j) * D_ + d0 + x];
    }
    __syncthreads();
    _Float16* hTb = hT + (size_t)b * D_ * N_;
#pragma unroll
    for (int k = 0; k < 4; ++k) {
        int d = y + 8 * k;
        hTb[(size_t)(d0 + d) * N_ + j0 + x] = (_Float16)tile[x][d];
    }
}

// ---------------------------------------------------------------------------
// Kernel 3: fused masked-softmax + prob@h + ELU.
// Block = 128 threads (4 waves) handles 16 rows of one batch.
// Pass A: row max (branchless, int4 adj loads).
// Pass B: 64-wide j chunks, double-buffered f16 prob tile in LDS (1 barrier /
//         chunk), 4x v_wmma_f32_16x16x32_f16 per chunk. Wave w owns output
//         columns [32w, 32w+32).
// ---------------------------------------------------------------------------
#define PA_STRIDE 72   // halves per row: 144B, 16B-multiple, bank-skewed

__device__ __forceinline__ float leaky(float s) {
    return fmaxf(s, ALPHA * s);   // exact LeakyReLU for both signs
}

__global__ __launch_bounds__(128) void gat_main(const int* __restrict__ adj,
                                                const float* __restrict__ f1,
                                                const float* __restrict__ f2,
                                                const _Float16* __restrict__ hT,
                                                float* __restrict__ out) {
    __shared__ float f2s[N_];
    __shared__ float f1s[16];
    __shared__ float rmax[16];
    __shared__ float rsum[16];
    __shared__ float rsumP[16][8];
    __shared__ __align__(16) _Float16 pA[2][16 * PA_STRIDE];

    const int tid  = threadIdx.x;
    const int lane = tid & 31;
    const int wave = tid >> 5;
    const int i0   = blockIdx.x * 16;
    const int b    = blockIdx.y;

    // ---- stage 0: f2 row of this batch + f1 of our 16 rows into LDS ----
    const float* f2b = f2 + (size_t)b * N_;
    for (int k = tid * 4; k < N_; k += 128 * 4)
        *reinterpret_cast<float4*>(&f2s[k]) =
            *reinterpret_cast<const float4*>(&f2b[k]);
    if (tid < 16) f1s[tid] = f1[(size_t)b * N_ + i0 + tid];
    __syncthreads();

    const int* adjb = adj + ((size_t)b * N_ + i0) * N_;

    // ---- pass A: per-row max of masked leaky(f1+f2) ----
#pragma unroll
    for (int rr = 0; rr < 4; ++rr) {
        int i = wave * 4 + rr;
        float fi = f1s[i];
        const int4* arow = reinterpret_cast<const int4*>(adjb + (size_t)i * N_);
        float m = -3.0e38f;
        for (int jv = lane; jv < N_ / 4; jv += 32) {
            int4 a = arow[jv];
            float4 fv = *reinterpret_cast<const float4*>(&f2s[jv * 4]);
            float s0 = leaky(fi + fv.x);
            float s1 = leaky(fi + fv.y);
            float s2 = leaky(fi + fv.z);
            float s3 = leaky(fi + fv.w);
            m = (a.x > 0) ? fmaxf(m, s0) : m;
            m = (a.y > 0) ? fmaxf(m, s1) : m;
            m = (a.z > 0) ? fmaxf(m, s2) : m;
            m = (a.w > 0) ? fmaxf(m, s3) : m;
        }
#pragma unroll
        for (int off = 16; off; off >>= 1) m = fmaxf(m, __shfl_xor(m, off));
        if (lane == 0) rmax[i] = m;
    }
    __syncthreads();

    // ---- pass B setup ----
    const int ip  = tid >> 3;          // row this thread produces probs for
    const int jc4 = (tid & 7) * 4;     // first of 4 contiguous j's (per half)
    const float fi = f1s[ip];
    const float mi = rmax[ip];
    const int* arow = adjb + (size_t)ip * N_;
    float sumAcc = 0.f;

    const int lm = lane & 15;          // N / M-within-half index
    const int lh = lane >> 4;          // half-wave
    const int d0 = wave * 32;
    const int d1 = d0 + 16;
    const _Float16* hTb = hT + (size_t)b * D_ * N_;
    const _Float16* hb0 = hTb + (size_t)(d0 + lm) * N_ + 16 * lh;
    const _Float16* hb1 = hTb + (size_t)(d1 + lm) * N_ + 16 * lh;

    v8f acc0 = {}; v8f acc1 = {};

    int buf = 0;
    for (int jb = 0; jb < N_; jb += 64, buf ^= 1) {
        _Float16* pAb = pA[buf];
        // cooperative prob tile: 16 x 64, f16, branchless
#pragma unroll
        for (int half = 0; half < 2; ++half) {
            int jc = jc4 + 32 * half;
            int j  = jb + jc;
            int4 a = *reinterpret_cast<const int4*>(arow + j);
            float4 fv = *reinterpret_cast<const float4*>(&f2s[j]);
            float p0 = __expf(leaky(fi + fv.x) - mi);
            float p1 = __expf(leaky(fi + fv.y) - mi);
            float p2 = __expf(leaky(fi + fv.z) - mi);
            float p3 = __expf(leaky(fi + fv.w) - mi);
            p0 = (a.x > 0) ? p0 : 0.0f;
            p1 = (a.y > 0) ? p1 : 0.0f;
            p2 = (a.z > 0) ? p2 : 0.0f;
            p3 = (a.w > 0) ? p3 : 0.0f;
            sumAcc += (p0 + p1) + (p2 + p3);
            v4h pk = { (_Float16)p0, (_Float16)p1, (_Float16)p2, (_Float16)p3 };
            *reinterpret_cast<v4h*>(&pAb[ip * PA_STRIDE + jc]) = pk;
        }
        __syncthreads();   // single barrier per chunk (double-buffered pA)

#pragma unroll
        for (int kk = 0; kk < 2; ++kk) {
            int jk = jb + 32 * kk;
            // A fragment: lane(lm,lh) holds K = 16lh + e, contiguous
            const _Float16* pArow = pAb + lm * PA_STRIDE + 32 * kk + 16 * lh;
            v8h alo = *reinterpret_cast<const v8h*>(pArow);
            v8h ahi = *reinterpret_cast<const v8h*>(pArow + 8);
            v16h afrag = __builtin_shufflevector(alo, ahi,
                0,1,2,3,4,5,6,7,8,9,10,11,12,13,14,15);

            // B fragments: lane(lm,lh) holds K = 16lh + e, contiguous in hT
            const _Float16* p0 = hb0 + jk;
            v8h b0lo = *reinterpret_cast<const v8h*>(p0);
            v8h b0hi = *reinterpret_cast<const v8h*>(p0 + 8);
            v16h bf0 = __builtin_shufflevector(b0lo, b0hi,
                0,1,2,3,4,5,6,7,8,9,10,11,12,13,14,15);
            const _Float16* p1 = hb1 + jk;
            v8h b1lo = *reinterpret_cast<const v8h*>(p1);
            v8h b1hi = *reinterpret_cast<const v8h*>(p1 + 8);
            v16h bf1 = __builtin_shufflevector(b1lo, b1hi,
                0,1,2,3,4,5,6,7,8,9,10,11,12,13,14,15);

            acc0 = __builtin_amdgcn_wmma_f32_16x16x32_f16(false, afrag, false, bf0,
                                                          (short)0, acc0, false, false);
            acc1 = __builtin_amdgcn_wmma_f32_16x16x32_f16(false, afrag, false, bf1,
                                                          (short)0, acc1, false, false);
        }
    }

    // ---- row sums ----
    rsumP[ip][tid & 7] = sumAcc;
    __syncthreads();
    if (tid < 16) {
        float rs = 0.f;
#pragma unroll
        for (int k = 0; k < 8; ++k) rs += rsumP[tid][k];
        rsum[tid] = rs;
    }
    __syncthreads();

    // ---- epilogue: normalize, ELU, store ----
    float* orow = out + ((size_t)b * N_ + i0) * D_;
#pragma unroll
    for (int r = 0; r < 8; ++r) {
        int M = r + 8 * lh;                 // C/D layout: VGPR r, half lh
        float inv = 1.0f / rsum[M];
        float v0 = acc0[r] * inv;
        v0 = v0 > 0.f ? v0 : (__expf(v0) - 1.0f);
        orow[(size_t)M * D_ + d0 + lm] = v0;
        float v1 = acc1[r] * inv;
        v1 = v1 > 0.f ? v1 : (__expf(v1) - 1.0f);
        orow[(size_t)M * D_ + d1 + lm] = v1;
    }
}

// ---------------------------------------------------------------------------
extern "C" void kernel_launch(void* const* d_in, const int* in_sizes, int n_in,
                              void* d_out, int out_size, void* d_ws, size_t ws_size,
                              hipStream_t stream) {
    const float* h   = (const float*)d_in[0];
    const int*   adj = (const int*)d_in[1];
    const float* a1  = (const float*)d_in[2];
    const float* a2  = (const float*)d_in[3];
    float* out = (float*)d_out;

    char* ws = (char*)d_ws;
    float*    f1  = (float*)ws;                                     // B*N floats
    float*    f2  = f1 + (size_t)B_ * N_;                           // B*N floats
    _Float16* hT  = (_Float16*)(ws + 2u * B_ * N_ * sizeof(float)); // B*D*N f16

    gat_f1f2<<<dim3(B_ * N_ / 4), dim3(128), 0, stream>>>(h, a1, a2, f1, f2);
    gat_transpose<<<dim3(N_ / 32, D_ / 32, B_), dim3(32, 8), 0, stream>>>(h, hT);
    gat_main<<<dim3(N_ / 16, B_), dim3(128), 0, stream>>>(adj, f1, f2, hT, out);
}
